// ImprovedCarbonGNN_13520557048011
// MI455X (gfx1250) — compile-verified, matrix operand order checked by reading, not run on hardware
//
#include <hip/hip_runtime.h>
#include <hip/hip_bf16.h>

// ---------------------------------------------------------------------------
// Problem constants (from reference)
// ---------------------------------------------------------------------------
#define NN 50000
#define EE 500000
#define IN_DIM 64
#define HID 256
#define OUTD 128
#define HEADS 8
#define DH 32
#define EPS 1e-5f

// ---------------------------------------------------------------------------
// bf16 helpers (bit-level, RTNE)
// ---------------------------------------------------------------------------
typedef __bf16 bf16_t;
typedef bf16_t v16bf __attribute__((ext_vector_type(16)));
typedef float  v8f   __attribute__((ext_vector_type(8)));

static __device__ __forceinline__ unsigned short f2bfbits(float f) {
    unsigned u = __float_as_uint(f);
    unsigned r = u + 0x7FFFu + ((u >> 16) & 1u);
    return (unsigned short)(r >> 16);
}
static __device__ __forceinline__ unsigned pk2(float a, float b) {
    return (unsigned)f2bfbits(a) | ((unsigned)f2bfbits(b) << 16);
}

// order-preserving float<->uint for atomic max
static __device__ __forceinline__ unsigned f2ord(float f) {
    unsigned u = __float_as_uint(f);
    return (u & 0x80000000u) ? ~u : (u | 0x80000000u);
}
static __device__ __forceinline__ float ord2f(unsigned u) {
    unsigned v = (u & 0x80000000u) ? (u & 0x7FFFFFFFu) : ~u;
    return __uint_as_float(v);
}

// ---------------------------------------------------------------------------
// Shared GEMM tile machinery
// Block: 256 threads (8 wave32). Tile: 128(M) x 128(N) x 32(K), double-buffered.
// LDS tiles stored as packed bf16 pairs in dwords, row stride 20 dwords (80B)
// -> all chunks 16B-aligned (ds_*_b128), conflict-free fragment reads.
// ---------------------------------------------------------------------------
#define TM 128
#define TN 128
#define TK 32
#define LSTRW 20   // dwords per LDS row (= 40 bf16 elements)

union frg16 { uint4 q[2]; v16bf v; };

static __device__ __forceinline__ void pack_store16(unsigned* dst, const float v[16]) {
    uint4 q0, q1;
    q0.x = pk2(v[0], v[1]);   q0.y = pk2(v[2], v[3]);
    q0.z = pk2(v[4], v[5]);   q0.w = pk2(v[6], v[7]);
    q1.x = pk2(v[8], v[9]);   q1.y = pk2(v[10], v[11]);
    q1.z = pk2(v[12], v[13]); q1.w = pk2(v[14], v[15]);
    ((uint4*)dst)[0] = q0;
    ((uint4*)dst)[1] = q1;
}

static __device__ __forceinline__ v16bf frag_load(const unsigned* rowbase, int kbw) {
    frg16 t;
    const uint4* p = (const uint4*)(rowbase + kbw);
    t.q[0] = p[0];
    t.q[1] = p[2];
    return t.v;
}

// ---------------------------------------------------------------------------
// WMMA GEMM: C[M,Nc] = op( A[M,K] @ W[K,Nc] + bias [+ addsrc] )
// Requires: K % 16 == 0 (alignment), Nc % 128 == 0.
// ---------------------------------------------------------------------------
__global__ __launch_bounds__(256) void wmma_gemm_kernel(
    const float* __restrict__ A, const float* __restrict__ W,
    const float* __restrict__ bias, const float* __restrict__ addsrc,
    float* __restrict__ C, int M, int K, int Nc, int relu)
{
    __shared__ __align__(16) unsigned As[2][TM * LSTRW];
    __shared__ __align__(16) unsigned Bs[2][TN * LSTRW];

    const int bm   = blockIdx.x * TM;
    const int bn   = blockIdx.y * TN;
    const int tid  = threadIdx.x;
    const int lane = tid & 31;
    const int wave = tid >> 5;
    const int wm   = (wave & 3) * 32;
    const int wn   = (wave >> 2) * 64;
    const int half = lane >> 4;
    const int l15  = lane & 15;
    const int kbw  = half * 4;
    const bool fullM = (bm + TM <= M);

    const int ar  = tid >> 1;            // A row in tile
    const int ac0 = (tid & 1) * 16;      // A col in tile (0/16)
    const int bnl = tid & 127;           // B column in tile
    const int bk0 = (tid >> 7) * 16;     // B k-base in tile (0/16)

    auto stageA = [&](unsigned* dstAs, int k0) {
        float v[16];
        const int gr = bm + ar;
        if (fullM && (k0 + TK <= K)) {
            const float4* ap = (const float4*)(A + (size_t)gr * K + (k0 + ac0));
            float4 x0 = ap[0], x1 = ap[1], x2 = ap[2], x3 = ap[3];
            v[0]=x0.x; v[1]=x0.y; v[2]=x0.z; v[3]=x0.w;
            v[4]=x1.x; v[5]=x1.y; v[6]=x1.z; v[7]=x1.w;
            v[8]=x2.x; v[9]=x2.y; v[10]=x2.z; v[11]=x2.w;
            v[12]=x3.x; v[13]=x3.y; v[14]=x3.z; v[15]=x3.w;
        } else {
#pragma unroll
            for (int i = 0; i < 16; ++i) {
                const int gk = k0 + ac0 + i;
                v[i] = (gr < M && gk < K) ? A[(size_t)gr * K + gk] : 0.f;
            }
        }
        pack_store16(&dstAs[ar * LSTRW + (ac0 >> 1)], v);
    };
    auto stageB = [&](unsigned* dstBs, int k0) {
        float v[16];
        if (k0 + TK <= K) {
#pragma unroll
            for (int i = 0; i < 16; ++i)
                v[i] = W[(size_t)(k0 + bk0 + i) * Nc + (bn + bnl)];
        } else {
#pragma unroll
            for (int i = 0; i < 16; ++i) {
                const int gk = k0 + bk0 + i;
                v[i] = (gk < K) ? W[(size_t)gk * Nc + (bn + bnl)] : 0.f;
            }
        }
        pack_store16(&dstBs[bnl * LSTRW + (bk0 >> 1)], v);
    };

    v8f acc[2][4] = {};
    const int nk = (K + TK - 1) / TK;

    stageA(As[0], 0);
    stageB(Bs[0], 0);
    __syncthreads();

    for (int kt = 0; kt < nk; ++kt) {
        const unsigned* curA = As[kt & 1];
        const unsigned* curB = Bs[kt & 1];
        // stage next tile into the other buffer while computing on this one
        if (kt + 1 < nk) {
            stageA(As[(kt + 1) & 1], (kt + 1) * TK);
            stageB(Bs[(kt + 1) & 1], (kt + 1) * TK);
        }

        v16bf afrag[2], bfrag[4];
#pragma unroll
        for (int tm = 0; tm < 2; ++tm)
            afrag[tm] = frag_load(&curA[(wm + tm * 16 + l15) * LSTRW], kbw);
#pragma unroll
        for (int tn = 0; tn < 4; ++tn)
            bfrag[tn] = frag_load(&curB[(wn + tn * 16 + l15) * LSTRW], kbw);

#pragma unroll
        for (int tm = 0; tm < 2; ++tm)
#pragma unroll
            for (int tn = 0; tn < 4; ++tn)
                acc[tm][tn] = __builtin_amdgcn_wmma_f32_16x16x32_bf16(
                    false, afrag[tm], false, bfrag[tn],
                    (short)0, acc[tm][tn], false, false);
        __syncthreads();
    }

    // ---- epilogue ----
    if (fullM) {
#pragma unroll
        for (int tm = 0; tm < 2; ++tm)
#pragma unroll
            for (int tn = 0; tn < 4; ++tn) {
                const int col = bn + wn + tn * 16 + l15;
                const float bv = bias ? bias[col] : 0.f;
#pragma unroll
                for (int j = 0; j < 8; ++j) {
                    const int row = bm + wm + tm * 16 + half * 8 + j;
                    const size_t off = (size_t)row * Nc + col;
                    float v = acc[tm][tn][j] + bv;
                    if (addsrc) v += addsrc[off];
                    if (relu) v = fmaxf(v, 0.f);
                    C[off] = v;
                }
            }
    } else {
#pragma unroll
        for (int tm = 0; tm < 2; ++tm)
#pragma unroll
            for (int tn = 0; tn < 4; ++tn) {
                const int col = bn + wn + tn * 16 + l15;
                const float bv = bias ? bias[col] : 0.f;
#pragma unroll
                for (int j = 0; j < 8; ++j) {
                    const int row = bm + wm + tm * 16 + half * 8 + j;
                    if (row < M) {
                        const size_t off = (size_t)row * Nc + col;
                        float v = acc[tm][tn][j] + bv;
                        if (addsrc) v += addsrc[off];
                        if (relu) v = fmaxf(v, 0.f);
                        C[off] = v;
                    }
                }
            }
    }
}

// ---------------------------------------------------------------------------
// Edge-gather WMMA GEMM: A row e = [node[src[e]](128) | node[dst[e]](128) | eattr(3)]
// K = 259 (zero-padded in the last K tile). C = ee @ W + bias.
// ---------------------------------------------------------------------------
__global__ __launch_bounds__(256) void wmma_edge_gemm_kernel(
    const float* __restrict__ node, const float* __restrict__ eattr,
    const int* __restrict__ src, const int* __restrict__ dst,
    const float* __restrict__ W, const float* __restrict__ bias,
    float* __restrict__ C, int M, int K, int Nc)
{
    __shared__ __align__(16) unsigned As[2][TM * LSTRW];
    __shared__ __align__(16) unsigned Bs[2][TN * LSTRW];

    const int bm   = blockIdx.x * TM;
    const int bn   = blockIdx.y * TN;
    const int tid  = threadIdx.x;
    const int lane = tid & 31;
    const int wave = tid >> 5;
    const int wm   = (wave & 3) * 32;
    const int wn   = (wave >> 2) * 64;
    const int half = lane >> 4;
    const int l15  = lane & 15;
    const int kbw  = half * 4;
    const bool fullM = (bm + TM <= M);

    const int ar  = tid >> 1;
    const int ac0 = (tid & 1) * 16;
    const int bnl = tid & 127;
    const int bk0 = (tid >> 7) * 16;

    // row gather indices (uniform across k-loop)
    const long e = (long)bm + ar;
    int gs = 0, gd = 0;
    if (e < M) { gs = src[e]; gd = dst[e]; }

    auto stageA = [&](unsigned* dstAs, int k0) {
        float v[16];
        if (fullM && (k0 + TK <= 2 * OUTD)) {
            // contiguous 16 floats from a node row
            const float* base = (k0 < OUTD)
                ? (node + (size_t)gs * OUTD + (k0 + ac0))
                : (node + (size_t)gd * OUTD + (k0 - OUTD + ac0));
            const float4* ap = (const float4*)base;
            float4 x0 = ap[0], x1 = ap[1], x2 = ap[2], x3 = ap[3];
            v[0]=x0.x; v[1]=x0.y; v[2]=x0.z; v[3]=x0.w;
            v[4]=x1.x; v[5]=x1.y; v[6]=x1.z; v[7]=x1.w;
            v[8]=x2.x; v[9]=x2.y; v[10]=x2.z; v[11]=x2.w;
            v[12]=x3.x; v[13]=x3.y; v[14]=x3.z; v[15]=x3.w;
        } else if (fullM) {
            // edge_attr tail tile: 3 valid columns, single divergent branch
#pragma unroll
            for (int i = 0; i < 16; ++i) v[i] = 0.f;
            if (ac0 == 0 && k0 == 2 * OUTD) {
                const float* ea = eattr + (size_t)e * 3;
                v[0] = ea[0]; v[1] = ea[1]; v[2] = ea[2];
            }
        } else {
#pragma unroll
            for (int i = 0; i < 16; ++i) {
                const int gk = k0 + ac0 + i;
                float x = 0.f;
                if (e < M) {
                    if (gk < OUTD)               x = node[(size_t)gs * OUTD + gk];
                    else if (gk < 2 * OUTD)      x = node[(size_t)gd * OUTD + (gk - OUTD)];
                    else if (gk < 2 * OUTD + 3)  x = eattr[(size_t)e * 3 + (gk - 2 * OUTD)];
                }
                v[i] = x;
            }
        }
        pack_store16(&dstAs[ar * LSTRW + (ac0 >> 1)], v);
    };
    auto stageB = [&](unsigned* dstBs, int k0) {
        float v[16];
        if (k0 + TK <= K) {
#pragma unroll
            for (int i = 0; i < 16; ++i)
                v[i] = W[(size_t)(k0 + bk0 + i) * Nc + (bn + bnl)];
        } else {
#pragma unroll
            for (int i = 0; i < 16; ++i) {
                const int gk = k0 + bk0 + i;
                v[i] = (gk < K) ? W[(size_t)gk * Nc + (bn + bnl)] : 0.f;
            }
        }
        pack_store16(&dstBs[bnl * LSTRW + (bk0 >> 1)], v);
    };

    v8f acc[2][4] = {};
    const int nk = (K + TK - 1) / TK;

    stageA(As[0], 0);
    stageB(Bs[0], 0);
    __syncthreads();

    for (int kt = 0; kt < nk; ++kt) {
        const unsigned* curA = As[kt & 1];
        const unsigned* curB = Bs[kt & 1];
        if (kt + 1 < nk) {
            stageA(As[(kt + 1) & 1], (kt + 1) * TK);
            stageB(Bs[(kt + 1) & 1], (kt + 1) * TK);
        }

        v16bf afrag[2], bfrag[4];
#pragma unroll
        for (int tm = 0; tm < 2; ++tm)
            afrag[tm] = frag_load(&curA[(wm + tm * 16 + l15) * LSTRW], kbw);
#pragma unroll
        for (int tn = 0; tn < 4; ++tn)
            bfrag[tn] = frag_load(&curB[(wn + tn * 16 + l15) * LSTRW], kbw);

#pragma unroll
        for (int tm = 0; tm < 2; ++tm)
#pragma unroll
            for (int tn = 0; tn < 4; ++tn)
                acc[tm][tn] = __builtin_amdgcn_wmma_f32_16x16x32_bf16(
                    false, afrag[tm], false, bfrag[tn],
                    (short)0, acc[tm][tn], false, false);
        __syncthreads();
    }

#pragma unroll
    for (int tm = 0; tm < 2; ++tm)
#pragma unroll
        for (int tn = 0; tn < 4; ++tn) {
            const int col = bn + wn + tn * 16 + l15;
            const float bv = bias ? bias[col] : 0.f;
#pragma unroll
            for (int j = 0; j < 8; ++j) {
                const long row = (long)bm + wm + tm * 16 + half * 8 + j;
                if (row < M) C[(size_t)row * Nc + col] = acc[tm][tn][j] + bv;
            }
        }
}

// ---------------------------------------------------------------------------
// Elementwise / reduction / scatter kernels
// ---------------------------------------------------------------------------
__global__ void zero_f_kernel(float* p, long n) {
    long i = (long)blockIdx.x * blockDim.x + threadIdx.x;
    if (i < n) p[i] = 0.f;
}
__global__ void zero_u_kernel(unsigned* p, long n) {
    long i = (long)blockIdx.x * blockDim.x + threadIdx.x;
    if (i < n) p[i] = 0u;
}
__global__ void bcast_bias_kernel(float* out, const float* bias, long n, int cols) {
    long i = (long)blockIdx.x * blockDim.x + threadIdx.x;
    if (i < n) out[i] = bias[i % cols];
}
__global__ void deg_kernel(const int* dst, float* deg, int e) {
    int i = blockIdx.x * blockDim.x + threadIdx.x;
    if (i < e) atomicAdd(&deg[dst[i]], 1.0f);
}
__global__ void bn_stats_kernel(const float* __restrict__ in, float* __restrict__ stats,
                                int M, int cols) {
    const int t = threadIdx.x;
    if (t >= cols) return;
    const int r0 = blockIdx.x * 512;
    float s = 0.f, q = 0.f;
    if (r0 + 512 <= M) {
        for (int i = 0; i < 512; ++i) {
            const float v = in[(size_t)(r0 + i) * cols + t];
            s += v; q += v * v;
        }
    } else {
        for (int i = 0; i < 512; ++i) {
            const int r = r0 + i;
            if (r < M) {
                const float v = in[(size_t)r * cols + t];
                s += v; q += v * v;
            }
        }
    }
    atomicAdd(&stats[t], s);
    atomicAdd(&stats[cols + t], q);
}
__global__ void bn_apply_kernel(const float* __restrict__ in, float* __restrict__ out,
                                const float* __restrict__ stats,
                                const float* __restrict__ gamma, const float* __restrict__ beta,
                                const float* __restrict__ res,
                                int M, int cols, int relu) {
    const long i = (long)blockIdx.x * blockDim.x + threadIdx.x;
    if (i >= (long)M * cols) return;
    const int c = (int)(i % cols);
    const float invM = 1.0f / (float)M;
    const float mean = stats[c] * invM;
    const float var  = stats[cols + c] * invM - mean * mean;
    float v = gamma[c] * (in[i] - mean) * rsqrtf(var + EPS) + beta[c];
    if (relu) v = fmaxf(v, 0.f);
    if (res) v += res[i];
    out[i] = v;
}
__global__ void gat_scores_kernel(const float* __restrict__ hh,
                                  const float* __restrict__ a_s, const float* __restrict__ a_d,
                                  float* ssrc, float* sdst, int n) {
    const int i = blockIdx.x * blockDim.x + threadIdx.x;
    if (i >= n * HEADS) return;
    const int nd = i >> 3, h = i & 7;
    const float* v = hh + (size_t)nd * HID + h * DH;
    float s1 = 0.f, s2 = 0.f;
#pragma unroll
    for (int j = 0; j < DH; ++j) {
        s1 += v[j] * a_s[h * DH + j];
        s2 += v[j] * a_d[h * DH + j];
    }
    ssrc[i] = s1; sdst[i] = s2;
}
__global__ void gat_alpha_kernel(const int* __restrict__ src, const int* __restrict__ dst,
                                 const float* __restrict__ ssrc, const float* __restrict__ sdst,
                                 float* alpha, unsigned* amax, int e) {
    const long i = (long)blockIdx.x * blockDim.x + threadIdx.x;
    if (i >= (long)e * HEADS) return;
    const long ed = i >> 3; const int h = (int)(i & 7);
    const int s = src[ed], d = dst[ed];
    float a = ssrc[s * HEADS + h] + sdst[d * HEADS + h];
    a = (a > 0.f) ? a : 0.2f * a;
    alpha[i] = a;
    atomicMax(&amax[d * HEADS + h], f2ord(a));
}
__global__ void gat_exp_kernel(const int* __restrict__ dst, float* __restrict__ alpha,
                               const unsigned* __restrict__ amax, float* denom, int e) {
    const long i = (long)blockIdx.x * blockDim.x + threadIdx.x;
    if (i >= (long)e * HEADS) return;
    const long ed = i >> 3; const int h = (int)(i & 7);
    const int d = dst[ed];
    const float ae = __expf(alpha[i] - ord2f(amax[d * HEADS + h]));
    alpha[i] = ae;
    atomicAdd(&denom[d * HEADS + h], ae);
}
__global__ void gat_scatter_kernel(const int* __restrict__ src, const int* __restrict__ dst,
                                   const float* __restrict__ hh, const float* __restrict__ ae,
                                   const float* __restrict__ denom, float* out) {
    const long e = blockIdx.x;
    const int f = threadIdx.x;
    const int s = src[e], d = dst[e];
    const int h = f >> 5;
    const float w = ae[e * HEADS + h] / denom[d * HEADS + h];
    atomicAdd(&out[(size_t)d * HID + f], hh[(size_t)s * HID + f] * w);
}
__global__ void sage_scatter_kernel(const int* __restrict__ src, const int* __restrict__ dst,
                                    const float* __restrict__ h, float* agg) {
    const long e = blockIdx.x;
    const int f = threadIdx.x;
    atomicAdd(&agg[(size_t)dst[e] * HID + f], h[(size_t)src[e] * HID + f]);
}
__global__ void sage_div_kernel(float* agg, const float* deg, long n) {
    const long i = (long)blockIdx.x * blockDim.x + threadIdx.x;
    if (i < n) agg[i] /= fmaxf(deg[i / HID], 1.0f);
}
__global__ void gemv_small_kernel(const float* __restrict__ A, const float* __restrict__ W,
                                  const float* __restrict__ bias, float* __restrict__ C,
                                  int M, int K, int Nc) {
    const long i = (long)blockIdx.x * blockDim.x + threadIdx.x;
    if (i >= (long)M * Nc) return;
    const int row = (int)(i / Nc), col = (int)(i % Nc);
    float acc = bias ? bias[col] : 0.f;
    const float* a = A + (size_t)row * K;
    for (int k = 0; k < K; ++k) acc += a[k] * W[k * Nc + col];
    C[i] = acc;
}
__global__ void gemv_wave_kernel(const float* __restrict__ A, const float* __restrict__ W,
                                 const float* __restrict__ bias, float* __restrict__ C,
                                 long M, int K) {
    const long row = (long)blockIdx.x * (blockDim.x >> 5) + (threadIdx.x >> 5);
    const int lane = threadIdx.x & 31;
    if (row >= M) return;
    const float* a = A + (size_t)row * K;
    float acc = 0.f;
    for (int k = lane; k < K; k += 32) acc += a[k] * W[k];
#pragma unroll
    for (int off = 16; off; off >>= 1) acc += __shfl_down(acc, off, 32);
    if (lane == 0) C[row] = acc + bias[0];
}

// ---------------------------------------------------------------------------
// Host orchestration
// ---------------------------------------------------------------------------
static inline dim3 g1(long n) { return dim3((unsigned)((n + 255) / 256)); }

extern "C" void kernel_launch(void* const* d_in, const int* in_sizes, int n_in,
                              void* d_out, int out_size, void* d_ws, size_t ws_size,
                              hipStream_t stream) {
    (void)in_sizes; (void)n_in; (void)out_size; (void)ws_size;
    const float* x     = (const float*)d_in[0];
    const int*   ei    = (const int*)d_in[1];
    const float* eattr = (const float*)d_in[2];
    int p = 3;
    const float* win  = (const float*)d_in[p++]; const float* bin  = (const float*)d_in[p++];
    const float* gin  = (const float*)d_in[p++]; const float* btin = (const float*)d_in[p++];
    const float* wgat = (const float*)d_in[p++];
    const float* asrc = (const float*)d_in[p++]; const float* adst = (const float*)d_in[p++];
    const float* bgat = (const float*)d_in[p++];
    const float* g0   = (const float*)d_in[p++]; const float* b0   = (const float*)d_in[p++];
    const float* wl1  = (const float*)d_in[p++]; const float* bl1  = (const float*)d_in[p++];
    const float* wr1  = (const float*)d_in[p++];
    const float* g1_  = (const float*)d_in[p++]; const float* b1_  = (const float*)d_in[p++];
    const float* wl2  = (const float*)d_in[p++]; const float* bl2  = (const float*)d_in[p++];
    const float* wr2  = (const float*)d_in[p++];
    const float* g2_  = (const float*)d_in[p++]; const float* b2_  = (const float*)d_in[p++];
    const float* wl3  = (const float*)d_in[p++]; const float* bl3  = (const float*)d_in[p++];
    const float* wr3  = (const float*)d_in[p++];
    const float* g3_  = (const float*)d_in[p++]; const float* b3_  = (const float*)d_in[p++];
    const float* cfw1 = (const float*)d_in[p++]; const float* cfb1 = (const float*)d_in[p++];
    const float* cfg  = (const float*)d_in[p++]; const float* cfbb = (const float*)d_in[p++];
    const float* cfw2 = (const float*)d_in[p++]; const float* cfb2 = (const float*)d_in[p++];
    const float* cfw3 = (const float*)d_in[p++]; const float* cfb3 = (const float*)d_in[p++];
    const float* scw1 = (const float*)d_in[p++]; const float* scb1 = (const float*)d_in[p++];
    const float* scg  = (const float*)d_in[p++]; const float* scbb = (const float*)d_in[p++];
    const float* scw2 = (const float*)d_in[p++]; const float* scb2 = (const float*)d_in[p++];
    const float* scw3 = (const float*)d_in[p++]; const float* scb3 = (const float*)d_in[p++];
    const float* locw = (const float*)d_in[p++]; const float* locb = (const float*)d_in[p++];
    const float* pfw  = (const float*)d_in[p++]; const float* pfb  = (const float*)d_in[p++];

    const int* src = ei;
    const int* dst = ei + EE;

    float* ws = (float*)d_ws;
    size_t off = 0;
    auto alloc = [&](size_t n) { float* q = ws + off; off += n; return q; };
    float* bufA = alloc((size_t)NN * HID);
    float* bufB = alloc((size_t)NN * HID);
    float* bufC = alloc((size_t)NN * HID);
    float* bufD = alloc((size_t)NN * HID);
    float* deg  = alloc(NN);
    float* ssrc = alloc((size_t)NN * HEADS);
    float* sdst = alloc((size_t)NN * HEADS);
    unsigned* amax = (unsigned*)alloc((size_t)NN * HEADS);
    float* denom = alloc((size_t)NN * HEADS);
    float* ae    = alloc((size_t)EE * HEADS);
    float* stats = alloc(2 * HID);
    float* z     = alloc((size_t)EE * HID);   // [E,256]
    float* z2    = alloc((size_t)EE * OUTD);  // [E,128]

    float* node   = (float*)d_out;                 // [N,128]
    float* carbon = node + (size_t)NN * OUTD;      // [E]
    float* sup    = carbon + EE;                   // [N,4]
    float* loc    = sup + (size_t)NN * 4;          // [N,3]
    float* perf   = loc + (size_t)NN * 3;          // [N,1]

    auto gemm = [&](const float* A, const float* W, const float* bias, const float* add,
                    float* C, int M, int K, int Nc, int relu) {
        dim3 grid((M + TM - 1) / TM, (Nc + TN - 1) / TN);
        wmma_gemm_kernel<<<grid, 256, 0, stream>>>(A, W, bias, add, C, M, K, Nc, relu);
    };
    auto bn = [&](const float* in, float* out, const float* g, const float* b,
                  const float* res, int M, int cols, int relu) {
        zero_f_kernel<<<g1(2 * cols), 256, 0, stream>>>(stats, 2 * cols);
        bn_stats_kernel<<<dim3((M + 511) / 512), 256, 0, stream>>>(in, stats, M, cols);
        bn_apply_kernel<<<g1((long)M * cols), 256, 0, stream>>>(in, out, stats, g, b, res, M, cols, relu);
    };

    // ---- degrees ----
    zero_f_kernel<<<g1(NN), 256, 0, stream>>>(deg, NN);
    deg_kernel<<<g1(EE), 256, 0, stream>>>(dst, deg, EE);

    // ---- input layer: h = relu(BN(x @ win + bin)) ----
    gemm(x, win, bin, nullptr, bufA, NN, IN_DIM, HID, 0);
    bn(bufA, bufA, gin, btin, nullptr, NN, HID, 1);                 // h = bufA

    // ---- GAT layer ----
    gemm(bufA, wgat, nullptr, nullptr, bufC, NN, HID, HID, 0);      // hh = bufC
    gat_scores_kernel<<<g1((long)NN * HEADS), 256, 0, stream>>>(bufC, asrc, adst, ssrc, sdst, NN);
    zero_u_kernel<<<g1((long)NN * HEADS), 256, 0, stream>>>(amax, (long)NN * HEADS);
    zero_f_kernel<<<g1((long)NN * HEADS), 256, 0, stream>>>(denom, (long)NN * HEADS);
    gat_alpha_kernel<<<g1((long)EE * HEADS), 256, 0, stream>>>(src, dst, ssrc, sdst, ae, amax, EE);
    gat_exp_kernel<<<g1((long)EE * HEADS), 256, 0, stream>>>(dst, ae, amax, denom, EE);
    bcast_bias_kernel<<<g1((long)NN * HID), 256, 0, stream>>>(bufD, bgat, (long)NN * HID, HID);
    gat_scatter_kernel<<<dim3(EE), 256, 0, stream>>>(src, dst, bufC, ae, denom, bufD);
    bn(bufD, bufB, g0, b0, bufA, NN, HID, 1);                       // h = bufB

    // ---- SAGE 1 ----
    zero_f_kernel<<<g1((long)NN * HID), 256, 0, stream>>>(bufC, (long)NN * HID);
    sage_scatter_kernel<<<dim3(EE), 256, 0, stream>>>(src, dst, bufB, bufC);
    sage_div_kernel<<<g1((long)NN * HID), 256, 0, stream>>>(bufC, deg, (long)NN * HID);
    gemm(bufB, wr1, nullptr, nullptr, bufD, NN, HID, HID, 0);
    gemm(bufC, wl1, bl1, bufD, bufA, NN, HID, HID, 0);
    bn(bufA, bufD, g1_, b1_, bufB, NN, HID, 1);                     // h = bufD

    // ---- SAGE 2 ----
    zero_f_kernel<<<g1((long)NN * HID), 256, 0, stream>>>(bufC, (long)NN * HID);
    sage_scatter_kernel<<<dim3(EE), 256, 0, stream>>>(src, dst, bufD, bufC);
    sage_div_kernel<<<g1((long)NN * HID), 256, 0, stream>>>(bufC, deg, (long)NN * HID);
    gemm(bufD, wr2, nullptr, nullptr, bufA, NN, HID, HID, 0);
    gemm(bufC, wl2, bl2, bufA, bufB, NN, HID, HID, 0);
    bn(bufB, bufA, g2_, b2_, bufD, NN, HID, 1);                     // h = bufA

    // ---- SAGE 3 (HID -> OUT) + final BN -> node ----
    zero_f_kernel<<<g1((long)NN * HID), 256, 0, stream>>>(bufC, (long)NN * HID);
    sage_scatter_kernel<<<dim3(EE), 256, 0, stream>>>(src, dst, bufA, bufC);
    sage_div_kernel<<<g1((long)NN * HID), 256, 0, stream>>>(bufC, deg, (long)NN * HID);
    gemm(bufA, wr3, nullptr, nullptr, bufB, NN, HID, OUTD, 0);
    gemm(bufC, wl3, bl3, bufB, bufD, NN, HID, OUTD, 0);
    bn(bufD, node, g3_, b3_, nullptr, NN, OUTD, 0);                 // node -> d_out

    // ---- edge (carbon) head ----
    {
        dim3 grid((EE + TM - 1) / TM, (HID + TN - 1) / TN);
        wmma_edge_gemm_kernel<<<grid, 256, 0, stream>>>(node, eattr, src, dst,
                                                        cfw1, cfb1, z, EE, 2 * OUTD + 3, HID);
    }
    bn(z, z, cfg, cfbb, nullptr, EE, HID, 1);
    gemm(z, cfw2, cfb2, nullptr, z2, EE, HID, OUTD, 1);
    gemv_wave_kernel<<<dim3((EE + 7) / 8), 256, 0, stream>>>(z2, cfw3, cfb3, carbon, EE, OUTD);

    // ---- supply head ----
    gemm(node, scw1, scb1, nullptr, bufA, NN, OUTD, HID, 0);
    bn(bufA, bufA, scg, scbb, nullptr, NN, HID, 1);
    gemm(bufA, scw2, scb2, nullptr, bufB, NN, HID, OUTD, 1);
    gemv_small_kernel<<<g1((long)NN * 4), 256, 0, stream>>>(bufB, scw3, scb3, sup, NN, OUTD, 4);

    // ---- loc / perf heads ----
    gemv_small_kernel<<<g1((long)NN * 3), 256, 0, stream>>>(node, locw, locb, loc, NN, OUTD, 3);
    gemv_wave_kernel<<<dim3((NN + 7) / 8), 256, 0, stream>>>(node, pfw, pfb, perf, NN, OUTD);
}
